// DefectPredictionGNN_6021544149482
// MI455X (gfx1250) — matrix operand order repaired and checked by reading, not compiled, not run on hardware
//
#include <hip/hip_runtime.h>
#include <hip/hip_bf16.h>

// ---------------------------------------------------------------------------
// CDNA5 (gfx1250) fused GNN inference.
// wave32; WMMA v_wmma_f32_16x16x32_bf16 for every non-trivial GEMM;
// GLOBAL_LOAD_ASYNC_TO_LDS_B128 + s_wait_asynccnt for head-kernel staging.
// ---------------------------------------------------------------------------

typedef __attribute__((ext_vector_type(16))) __bf16 v16bf;
typedef __attribute__((ext_vector_type(8)))  float  v8f;
typedef __attribute__((ext_vector_type(4)))  float  f32x4;

union FragBF {
    v16bf v;
    f32x4 q[2];
};

// One 16x16 output tile: D += A(16 x nk*32) * B(nk*32 x 16), bf16 in / f32 acc.
// aRow : per-lane pointer to row (M = lane&15) of the A tile, K contiguous.
// bCol : per-lane pointer to column (N = lane&15) of B, K contiguous (stride nk*32).
// Fragment layouts follow CDNA5 ISA 7.12.2 (16-bit A 16x32; B 32x16).
__device__ __forceinline__ v8f mm_tile(const __bf16* __restrict__ aRow,
                                       const __bf16* __restrict__ bCol,
                                       int lane, int nk) {
    v8f acc = {};
    const int aSel = (lane & 16) ? 8  : 0;   // K sub-block select for A
    const int bSel = (lane & 16) ? 8  : 0;   // K sub-block select for B
    for (int kc = 0; kc < nk; ++kc) {
        FragBF a, b;
        const __bf16* ap = aRow + kc * 32 + aSel;
        a.q[0] = *(const f32x4*)(ap);        // K = base+0..7   (or 8..15)
        a.q[1] = *(const f32x4*)(ap + 16);   // K = base+16..23 (or 24..31)
        const __bf16* bp = bCol + kc * 32 + bSel;
        b.q[0] = *(const f32x4*)(bp);        // K = base+0..7   (or 8..15)
        b.q[1] = *(const f32x4*)(bp + 16);   // K = base+16..23 (or 24..31)
        acc = __builtin_amdgcn_wmma_f32_16x16x32_bf16(
            false, a.v, false, b.v, (short)0, acc, false, false);
    }
    return acc;
}

// Async copy of one 16-byte chunk (per lane) from global memory to LDS.
// ldsAddr: LDS byte address (generic LDS pointer truncated to 32 bits).
// Tracked with ASYNCcnt; waited with s_wait_asynccnt.
__device__ __forceinline__ void async_copy_b128(unsigned ldsAddr, const void* gsrc) {
    unsigned long long ga = (unsigned long long)gsrc;
    asm volatile("global_load_async_to_lds_b128 %0, %1, off"
                 :: "v"(ldsAddr), "v"(ga) : "memory");
}

__device__ __forceinline__ void wait_asynccnt0() {
    asm volatile("s_wait_asynccnt 0" ::: "memory");
}

// ---------------------------------------------------------------------------
// Kernel 0: weight convert  fp32 [K][N] row-major -> bf16 [N][Kpad] (K-major)
// ---------------------------------------------------------------------------
struct ConvertArgs {
    const float* src[7];
    __bf16*      dst[7];
    int K[7], N[7], KP[7];
};

__global__ __launch_bounds__(256) void convert_weights(ConvertArgs C) {
    const int m = blockIdx.x;
    const float* s = C.src[m];
    __bf16*      d = C.dst[m];
    const int K = C.K[m], N = C.N[m], KP = C.KP[m];
    const int total = N * KP;
    for (int i = threadIdx.x; i < total; i += blockDim.x) {
        const int n = i / KP, k = i - n * KP;
        d[(long)n * KP + k] = (k < K) ? (__bf16)s[(long)k * N + n] : (__bf16)0.f;
    }
}

// ---------------------------------------------------------------------------
// Kernel 1: fused conv1+bn1+relu -> conv2+bn2+relu -> conv3+bn3+relu -> pool
// block = 128 threads (4 waves), 8 graphs (64 node rows) per block.
// ---------------------------------------------------------------------------
struct ConvArgs {
    const float* x;
    const __bf16 *W1, *W2, *W3;              // bf16 weights in ws (K-major)
    const float *cb1, *cb2, *cb3;            // conv biases
    const float *g1, *b1, *m1, *v1;
    const float *g2, *b2, *m2, *v2;
    const float *g3, *b3, *m3, *v3;
    __bf16* G;                               // pooled g vectors [B][288]
};

#define RS2 0.70710678f   // 1/sqrt(2)
#define RS3 0.57735027f   // 1/sqrt(3)

__global__ __launch_bounds__(128) void gnn_conv_kernel(ConvArgs A) {
    __shared__ __bf16 sX[64][32];     // padded bf16 input tile
    __shared__ __bf16 sH[64][128];    // bf16 activations (A operand)
    __shared__ float  sXW[64][128];   // fp32 XW staging for aggregation

    const int tid  = threadIdx.x;
    const int lane = tid & 31;
    const int wv   = tid >> 5;
    const long R0  = (long)blockIdx.x * 64;   // first node row of this block

    // stage x -> bf16, zero-pad K to 32
    for (int e = 0; e < 16; ++e) {
        const int idx = e * 128 + tid;        // 64*32 = 2048 slots
        const int r = idx >> 5, c = idx & 31;
        float val = (c < 11) ? A.x[(R0 + r) * 11 + c] : 0.f;
        sX[r][c] = (__bf16)val;
    }
    __syncthreads();

    // Per-layer: WMMA GEMM into sXW, then chain-stencil aggregation + BN + ReLU
#define CONV_LAYER(AROWBASE, NK, WPTR, CBIAS, BNG, BNB, BNM, BNV)              \
    {                                                                          \
        const __bf16* aRow_ = &(AROWBASE)[wv * 16 + (lane & 15)][0];           \
        for (int nt = 0; nt < 8; ++nt) {                                       \
            const int col = nt * 16 + (lane & 15);                             \
            v8f acc = mm_tile(aRow_, (WPTR) + (long)col * ((NK) * 32),         \
                              lane, (NK));                                     \
            const int rb = wv * 16 + ((lane & 16) ? 8 : 0);                    \
            for (int r = 0; r < 8; ++r) sXW[rb + r][col] = acc[r];             \
        }                                                                      \
        __syncthreads();                                                       \
        {                                                                      \
            const int c = tid;  /* 128 threads == 128 cols */                  \
            const float bia = (CBIAS)[c];                                      \
            const float sg = (BNG)[c], sb = (BNB)[c], sm = (BNM)[c];           \
            const float sv = rsqrtf((BNV)[c] + 1e-5f);                         \
            for (int r = 0; r < 64; ++r) {                                     \
                const int p = r & 7;                                           \
                const float dp = (p == 0 || p == 7) ? RS2 : RS3;               \
                float ag = dp * dp * sXW[r][c] + bia;                          \
                if (p > 0) ag += ((p == 1) ? RS2 : RS3) * dp * sXW[r - 1][c];  \
                if (p < 7) ag += ((p == 6) ? RS2 : RS3) * dp * sXW[r + 1][c];  \
                const float hh = sg * (ag - sm) * sv + sb;                     \
                sH[r][c] = (__bf16)fmaxf(hh, 0.f);                             \
            }                                                                  \
        }                                                                      \
        __syncthreads();                                                       \
    }

    CONV_LAYER(sX, 1, A.W1, A.cb1, A.g1, A.b1, A.m1, A.v1)   // 11(->32) -> 128
    CONV_LAYER(sH, 4, A.W2, A.cb2, A.g2, A.b2, A.m2, A.v2)   // 128 -> 128
    CONV_LAYER(sH, 4, A.W3, A.cb3, A.g3, A.b3, A.m3, A.v3)   // 128 -> 128
#undef CONV_LAYER

    // pooling: per graph max/mean over 8 rows -> g[b][0:128]=max, [128:256]=mean
    for (int t = 0; t < 8; ++t) {
        const int task = t * 128 + tid;        // 8 graphs * 128 cols
        const int gl = task >> 7, c = task & 127;
        float mx = -3.4e38f, sm = 0.f;
        for (int r = 0; r < 8; ++r) {
            const float hv = (float)sH[gl * 8 + r][c];
            mx = fmaxf(mx, hv);
            sm += hv;
        }
        __bf16* grow = A.G + ((long)blockIdx.x * 8 + gl) * 288;
        grow[c]       = (__bf16)mx;
        grow[128 + c] = (__bf16)(sm * 0.125f);
    }
    // raw_x: first 3 features of each of the 8 nodes -> cols 256..279
    if (tid < 192) {
        const int gl = tid / 24, idx = tid - gl * 24;
        const int s = idx / 3, j = idx - s * 3;
        const long node = R0 + gl * 8 + s;
        A.G[((long)blockIdx.x * 8 + gl) * 288 + 256 + idx] = (__bf16)A.x[node * 11 + j];
    }
    // zero pad cols 280..287
    if (tid < 64) {
        const int gl = tid >> 3, k = tid & 7;
        A.G[((long)blockIdx.x * 8 + gl) * 288 + 280 + k] = (__bf16)0.f;
    }
}

// ---------------------------------------------------------------------------
// Kernel 2: MLP heads. block = 64 threads (2 waves), 32 graphs per block.
// g rows are staged once into LDS with async b128 copies (read 3x by the
// three head layer-0 GEMMs).
// ---------------------------------------------------------------------------
struct HeadArgs {
    const __bf16 *G, *Wt0, *Wt1, *Wl0, *Ws0;
    const float *bt0, *bt1, *bt2, *wt2;
    const float *bl0, *bl1, *wl1;
    const float *bs0, *bs1, *ws1;
    float* out;
    int B;
};

__global__ __launch_bounds__(64) void head_kernel(HeadArgs A) {
    __shared__ __bf16 sG[32][288];   // staged g rows (async copy target), 18 KB
    __shared__ __bf16 sA[32][128];   // bf16 intermediate (A operand for layer 1)
    __shared__ float  sF[32][64];    // fp32 staging for tiny VALU final layers

    const int tid  = threadIdx.x;    // 0..63
    const int lane = tid & 31;
    const int wv   = tid >> 5;       // 0..1
    const long rowBase = (long)blockIdx.x * 32;

    // ---- async stage: 32 contiguous g rows (32*288 bf16 = 18432 B) -> sG ----
    {
        const char* gsrc = (const char*)(A.G + rowBase * 288);
        const unsigned ldsBase = (unsigned)(unsigned long long)(&sG[0][0]);
        for (int t = 0; t < 18; ++t) {
            const int chunk = t * 64 + tid;          // 1152 chunks of 16 B
            async_copy_b128(ldsBase + chunk * 16, gsrc + chunk * 16);
        }
        wait_asynccnt0();
    }
    __syncthreads();

    const __bf16* gRow = &sG[wv * 16 + (lane & 15)][0];
    const int rb = wv * 16 + ((lane & 16) ? 8 : 0);

    // ---- type head ----
    // layer0: 288 -> 128, relu
    for (int nt = 0; nt < 8; ++nt) {
        const int col = nt * 16 + (lane & 15);
        v8f acc = mm_tile(gRow, A.Wt0 + (long)col * 288, lane, 9);
        const float bia = A.bt0[col];
        for (int r = 0; r < 8; ++r)
            sA[rb + r][col] = (__bf16)fmaxf(acc[r] + bia, 0.f);
    }
    __syncthreads();
    // layer1: 128 -> 64, relu
    {
        const __bf16* aRow = &sA[wv * 16 + (lane & 15)][0];
        for (int nt = 0; nt < 4; ++nt) {
            const int col = nt * 16 + (lane & 15);
            v8f acc = mm_tile(aRow, A.Wt1 + (long)col * 128, lane, 4);
            const float bia = A.bt1[col];
            for (int r = 0; r < 8; ++r)
                sF[rb + r][col] = fmaxf(acc[r] + bia, 0.f);
        }
    }
    __syncthreads();
    // layer2: 64 -> 6 (VALU), logits
    for (int t = 0; t < 3; ++t) {
        const int task = t * 64 + tid;         // 32 rows * 6 cols = 192
        const int r = task / 6, c = task - r * 6;
        float s = A.bt2[c];
        for (int k = 0; k < 64; ++k) s += sF[r][k] * A.wt2[k * 6 + c];
        A.out[(rowBase + r) * 6 + c] = s;
    }
    __syncthreads();

    // ---- loc head ----
    // layer0: 288 -> 64, relu
    for (int nt = 0; nt < 4; ++nt) {
        const int col = nt * 16 + (lane & 15);
        v8f acc = mm_tile(gRow, A.Wl0 + (long)col * 288, lane, 9);
        const float bia = A.bl0[col];
        for (int r = 0; r < 8; ++r)
            sF[rb + r][col] = fmaxf(acc[r] + bia, 0.f);
    }
    __syncthreads();
    // layer1: 64 -> 2, sigmoid
    {
        const int r = tid >> 1, c = tid & 1;   // 32 rows * 2 cols = 64
        float s = A.bl1[c];
        for (int k = 0; k < 64; ++k) s += sF[r][k] * A.wl1[k * 2 + c];
        A.out[(long)A.B * 6 + (rowBase + r) * 2 + c] = 1.f / (1.f + __expf(-s));
    }
    __syncthreads();

    // ---- sev head ----
    // layer0: 288 -> 32, relu
    for (int nt = 0; nt < 2; ++nt) {
        const int col = nt * 16 + (lane & 15);
        v8f acc = mm_tile(gRow, A.Ws0 + (long)col * 288, lane, 9);
        const float bia = A.bs0[col];
        for (int r = 0; r < 8; ++r)
            sF[rb + r][col] = fmaxf(acc[r] + bia, 0.f);
    }
    __syncthreads();
    // layer1: 32 -> 1, sigmoid
    if (tid < 32) {
        float s = A.bs1[0];
        for (int k = 0; k < 32; ++k) s += sF[tid][k] * A.ws1[k];
        A.out[(long)A.B * 8 + rowBase + tid] = 1.f / (1.f + __expf(-s));
    }
}

// ---------------------------------------------------------------------------
// Input ordering (jax tree_leaves: dict keys sorted):
//  0:x 1:edge_index 2:batch
//  3-6   bn1 {b,g,m,v}   7-10 bn2   11-14 bn3
//  15,16 conv1 {b,w}     17,18 conv2   19,20 conv3
//  21,22 loc[0]{b,w}  23,24 loc[1]{b,w}
//  25,26 sev[0]{b,w}  27,28 sev[1]{b,w}
//  29,30 type[0]{b,w} 31,32 type[1]{b,w} 33,34 type[2]{b,w}
// ---------------------------------------------------------------------------
extern "C" void kernel_launch(void* const* d_in, const int* in_sizes, int n_in,
                              void* d_out, int out_size, void* d_ws, size_t ws_size,
                              hipStream_t stream) {
    (void)n_in; (void)out_size; (void)ws_size;
    const float* x = (const float*)d_in[0];
    const int B = in_sizes[0] / 88;   // N*11 / (8*11)

    const float *bn1b = (const float*)d_in[3],  *bn1g = (const float*)d_in[4];
    const float *bn1m = (const float*)d_in[5],  *bn1v = (const float*)d_in[6];
    const float *bn2b = (const float*)d_in[7],  *bn2g = (const float*)d_in[8];
    const float *bn2m = (const float*)d_in[9],  *bn2v = (const float*)d_in[10];
    const float *bn3b = (const float*)d_in[11], *bn3g = (const float*)d_in[12];
    const float *bn3m = (const float*)d_in[13], *bn3v = (const float*)d_in[14];
    const float *c1b = (const float*)d_in[15], *c1w = (const float*)d_in[16];
    const float *c2b = (const float*)d_in[17], *c2w = (const float*)d_in[18];
    const float *c3b = (const float*)d_in[19], *c3w = (const float*)d_in[20];
    const float *l0b = (const float*)d_in[21], *l0w = (const float*)d_in[22];
    const float *l1b = (const float*)d_in[23], *l1w = (const float*)d_in[24];
    const float *s0b = (const float*)d_in[25], *s0w = (const float*)d_in[26];
    const float *s1b = (const float*)d_in[27], *s1w = (const float*)d_in[28];
    const float *t0b = (const float*)d_in[29], *t0w = (const float*)d_in[30];
    const float *t1b = (const float*)d_in[31], *t1w = (const float*)d_in[32];
    const float *t2b = (const float*)d_in[33], *t2w = (const float*)d_in[34];

    // workspace layout (bytes)
    char* ws = (char*)d_ws;
    __bf16* W1  = (__bf16*)(ws + 0);        // [128][32]
    __bf16* W2  = (__bf16*)(ws + 8192);     // [128][128]
    __bf16* W3  = (__bf16*)(ws + 40960);    // [128][128]
    __bf16* Wt0 = (__bf16*)(ws + 73728);    // [128][288]
    __bf16* Wt1 = (__bf16*)(ws + 147456);   // [64][128]
    __bf16* Wl0 = (__bf16*)(ws + 163840);   // [64][288]
    __bf16* Ws0 = (__bf16*)(ws + 200704);   // [32][288]
    __bf16* G   = (__bf16*)(ws + 219136);   // [B][288]

    ConvertArgs CA;
    CA.src[0] = c1w; CA.dst[0] = W1;  CA.K[0] = 11;  CA.N[0] = 128; CA.KP[0] = 32;
    CA.src[1] = c2w; CA.dst[1] = W2;  CA.K[1] = 128; CA.N[1] = 128; CA.KP[1] = 128;
    CA.src[2] = c3w; CA.dst[2] = W3;  CA.K[2] = 128; CA.N[2] = 128; CA.KP[2] = 128;
    CA.src[3] = t0w; CA.dst[3] = Wt0; CA.K[3] = 280; CA.N[3] = 128; CA.KP[3] = 288;
    CA.src[4] = t1w; CA.dst[4] = Wt1; CA.K[4] = 128; CA.N[4] = 64;  CA.KP[4] = 128;
    CA.src[5] = l0w; CA.dst[5] = Wl0; CA.K[5] = 280; CA.N[5] = 64;  CA.KP[5] = 288;
    CA.src[6] = s0w; CA.dst[6] = Ws0; CA.K[6] = 280; CA.N[6] = 32;  CA.KP[6] = 288;
    convert_weights<<<7, 256, 0, stream>>>(CA);

    ConvArgs A;
    A.x = x;
    A.W1 = W1; A.W2 = W2; A.W3 = W3;
    A.cb1 = c1b; A.cb2 = c2b; A.cb3 = c3b;
    A.g1 = bn1g; A.b1 = bn1b; A.m1 = bn1m; A.v1 = bn1v;
    A.g2 = bn2g; A.b2 = bn2b; A.m2 = bn2m; A.v2 = bn2v;
    A.g3 = bn3g; A.b3 = bn3b; A.m3 = bn3m; A.v3 = bn3v;
    A.G = G;
    gnn_conv_kernel<<<B / 8, 128, 0, stream>>>(A);

    HeadArgs HA;
    HA.G = G; HA.Wt0 = Wt0; HA.Wt1 = Wt1; HA.Wl0 = Wl0; HA.Ws0 = Ws0;
    HA.bt0 = t0b; HA.bt1 = t1b; HA.bt2 = t2b; HA.wt2 = t2w;
    HA.bl0 = l0b; HA.bl1 = l1b; HA.wl1 = l1w;
    HA.bs0 = s0b; HA.bs1 = s1b; HA.ws1 = s1w;
    HA.out = (float*)d_out; HA.B = B;
    head_kernel<<<B / 32, 64, 0, stream>>>(HA);
}
